// RNNEncoder_5643587027287
// MI455X (gfx1250) — matrix operand order repaired and checked by reading, not compiled, not run on hardware
//
#include <hip/hip_runtime.h>
#include <hip/hip_bf16.h>

typedef __attribute__((ext_vector_type(16))) __bf16 v16bf;
typedef __attribute__((ext_vector_type(8)))  __bf16 v8bf;
typedef __attribute__((ext_vector_type(8)))  float  v8f;
typedef __attribute__((ext_vector_type(4)))  float  v4f;
typedef __attribute__((ext_vector_type(2)))  float  v2f;

#define HID   128
#define G3    384
#define BDIM  256          // 8 waves of 32, one 16-row tile per wave
#define TSTEPS 24
#define NSENS 512

// workspace layout (bytes): prepped by prep_kernel
#define WS_WT    0                     // 3 * 384*128 bf16 = 294912 (WhT0|WxT1|WhT1)
#define WS_EPI0  294912                // 384 * float4 = 6144
#define WS_EPI1  301056                // 384 * float2 = 3072
#define WS_BYTES 304128

// LDS layout (bytes)
#define OFF_W     0                    // 294912: bf16 weights [col][k]
#define OFF_STAGE 294912               // 8 waves * 16*128 bf16 = 32768
#define SMEM_BYTES 327680              // exactly 320 KB

#if __has_builtin(__builtin_amdgcn_tanhf)
#define FTANH(x) __builtin_amdgcn_tanhf(x)   // gfx1250 v_tanh_f32
#else
static __device__ __forceinline__ float FTANH(float v) {
  float e = __expf(-2.0f * v);
  return (1.0f - e) * __builtin_amdgcn_rcpf(1.0f + e);
}
#endif

static __device__ __forceinline__ v2f pair(const v8f& v, int g2) {
  v2f r; r.x = v[g2]; r.y = v[g2 + 1]; return r;
}
static __device__ __forceinline__ v2f splat2(float s) { v2f r; r.x = s; r.y = s; return r; }
static __device__ __forceinline__ v2f tanh2(v2f v) {
  v2f r; r.x = FTANH(v.x); r.y = FTANH(v.y); return r;
}
static __device__ __forceinline__ v2f sig2(v2f v) {   // sigmoid(v) = 0.5*tanh(v/2)+0.5
  return tanh2(v * 0.5f) * 0.5f + 0.5f;
}

#define WMMA_BF16(A, B, C) \
  __builtin_amdgcn_wmma_f32_16x16x32_bf16(false, (A), false, (B), (short)0, (C), false, false)

// -------- prep: transpose weights to bf16 [col][k] + pack epilogue tables --------
extern "C" __global__ void gru_prep_kernel(
    const float* __restrict__ Wx0, const float* __restrict__ Wh0,
    const float* __restrict__ bx0, const float* __restrict__ bh0,
    const float* __restrict__ Wx1, const float* __restrict__ Wh1,
    const float* __restrict__ bx1, const float* __restrict__ bh1,
    __bf16* __restrict__ wT, v4f* __restrict__ epi0g, v2f* __restrict__ epi1g) {
  const int i = blockIdx.x * blockDim.x + threadIdx.x;
  if (i < G3 * HID) {
    const int col = i >> 7, k = i & (HID - 1);
    wT[i]                 = (__bf16)Wh0[k * G3 + col];
    wT[G3 * HID + i]      = (__bf16)Wx1[k * G3 + col];
    wT[2 * G3 * HID + i]  = (__bf16)Wh1[k * G3 + col];
  }
  if (i < G3) {
    v4f e; e.x = Wx0[i]; e.y = Wx0[G3 + i]; e.z = bx0[i]; e.w = bh0[i];
    epi0g[i] = e;
    v2f e1; e1.x = bx1[i]; e1.y = bh1[i];
    epi1g[i] = e1;
  }
}

// Stage a 16x128 f32 C/D-layout tile into per-wave LDS as bf16 [row][k], then
// pull it back as four 16x32 A-fragments (contiguous k = 16*half + s, matching B).
static __device__ __forceinline__ void stage_tile(__bf16* stage, const v8f* h,
                                                  int ln, int hiw, v16bf* A) {
  #pragma unroll
  for (int nt = 0; nt < 8; ++nt)
    #pragma unroll
    for (int g = 0; g < 8; ++g)
      stage[(g + hiw * 8) * HID + nt * 16 + ln] = (__bf16)h[nt][g];
  #pragma unroll
  for (int kk = 0; kk < 4; ++kk) {
    const v8bf* p = (const v8bf*)(stage + ln * HID + kk * 32 + hiw * 16);
    const v8bf lo = p[0], hh = p[1];
    #pragma unroll
    for (int i = 0; i < 8; ++i) { A[kk][i] = lo[i]; A[kk][8 + i] = hh[i]; }
  }
}

// One 16x16x128 GEMM block-column triple (r,z,n gates) from LDS weights.
static __device__ __forceinline__ void gemm3(const __bf16* W, const v16bf* A,
                                             int cb, int ln, int hiw,
                                             v8f& gr, v8f& gz, v8f& gn) {
  #pragma unroll
  for (int kk = 0; kk < 4; ++kk) {
    const int ko = kk * 32 + hiw * 16;
    const v16bf br = *(const v16bf*)(W + (0 * 128 + cb + ln) * HID + ko);
    const v16bf bz = *(const v16bf*)(W + (1 * 128 + cb + ln) * HID + ko);
    const v16bf bn = *(const v16bf*)(W + (2 * 128 + cb + ln) * HID + ko);
    gr = WMMA_BF16(A[kk], br, gr);
    gz = WMMA_BF16(A[kk], bz, gz);
    gn = WMMA_BF16(A[kk], bn, gn);
  }
}

extern "C" __global__ void __launch_bounds__(BDIM)
gru_fused_kernel(const float* __restrict__ x, const __bf16* __restrict__ wT,
                 const v4f* __restrict__ epi0g, const v2f* __restrict__ epi1g,
                 float* __restrict__ out) {
  extern __shared__ char smem[];
  __bf16* WhT0 = (__bf16*)(smem + OFF_W);
  __bf16* WxT1 = WhT0 + G3 * HID;
  __bf16* WhT1 = WhT0 + 2 * G3 * HID;

  const int tid = threadIdx.x;

  // coalesced copy of prepped bf16 weights into LDS (18432 x 16B)
  {
    const v4f* src = (const v4f*)wT;
    v4f* dst = (v4f*)(smem + OFF_W);
    #pragma unroll 1
    for (int i = tid; i < (3 * G3 * HID * 2) / 16; i += BDIM) dst[i] = src[i];
  }
  __syncthreads();

  const int lane = tid & 31;
  const int ln   = lane & 15;
  const int hiw  = (lane >> 4) & 1;
  const int wv   = tid >> 5;
  __bf16* stage = (__bf16*)(smem + OFF_STAGE) + wv * (16 * HID);

  // persistent hidden state, C/D fragment layout
  v8f h0a[8], h1a[8];
  #pragma unroll
  for (int nt = 0; nt < 8; ++nt) { h0a[nt] = {}; h1a[nt] = {}; }

  const int M0 = blockIdx.x * 128 + wv * 16;   // this wave's 16 rows

  #pragma unroll 1
  for (int t = 0; t < TSTEPS; ++t) {
    // per-row x_t values (row = ln; duplicated across half-waves), packed in pairs
    const int Mx = M0 + ln;
    const int bb = Mx >> 9, sn = Mx & (NSENS - 1);
    const float* xp = x + (((bb * TSTEPS + t) * 2) * NSENS + sn);
    const float xa0 = xp[0];
    const float xa1 = xp[NSENS];
    v2f x0p[4], x1p[4];
    #pragma unroll
    for (int g = 0; g < 8; ++g) {
      const int src = (g + hiw * 8) * 4;
      const float r0 = __int_as_float(__builtin_amdgcn_ds_bpermute(src, __float_as_int(xa0)));
      const float r1 = __int_as_float(__builtin_amdgcn_ds_bpermute(src, __float_as_int(xa1)));
      if (g & 1) { x0p[g >> 1].y = r0; x1p[g >> 1].y = r1; }
      else       { x0p[g >> 1].x = r0; x1p[g >> 1].x = r1; }
    }

    // ================= layer 0 =================
    v16bf A0[4];
    stage_tile(stage, h0a, ln, hiw, A0);

    #pragma unroll
    for (int nt = 0; nt < 8; ++nt) {
      v8f ar = {}, az = {}, an = {};
      const int cb = nt * 16;
      gemm3(WhT0, A0, cb, ln, hiw, ar, az, an);
      const v4f er = epi0g[0 * 128 + cb + ln];
      const v4f ez = epi0g[1 * 128 + cb + ln];
      const v4f en = epi0g[2 * 128 + cb + ln];
      #pragma unroll
      for (int g2 = 0; g2 < 8; g2 += 2) {
        const int q = g2 >> 1;
        const v2f vr = x0p[q] * er.x + x1p[q] * er.y + splat2(er.z + er.w) + pair(ar, g2);
        const v2f vz = x0p[q] * ez.x + x1p[q] * ez.y + splat2(ez.z + ez.w) + pair(az, g2);
        const v2f rr = sig2(vr);
        const v2f zz = sig2(vz);
        const v2f vn = x0p[q] * en.x + x1p[q] * en.y + splat2(en.z) +
                       rr * (pair(an, g2) + splat2(en.w));
        const v2f nv = tanh2(vn);
        const v2f hc = pair(h0a[nt], g2);
        const v2f hn2 = nv + zz * (hc - nv);       // (1-z)*n + z*h
        h0a[nt][g2] = hn2.x; h0a[nt][g2 + 1] = hn2.y;
      }
    }

    // ================= layer 1 (two passes to cap register pressure) =================
    v16bf AI[4], AH[4];
    stage_tile(stage, h0a, ln, hiw, AI);
    stage_tile(stage, h1a, ln, hiw, AH);

    #pragma unroll
    for (int nt = 0; nt < 8; ++nt) {
      const int cb = nt * 16;
      // pass 1: gi1 = h0_new x Wx1  (AI live, AH idle)
      v8f ir = {}, iz = {}, inn = {};
      gemm3(WxT1, AI, cb, ln, hiw, ir, iz, inn);
      // pass 2: gh1 = h1 x Wh1  (AI dead after last nt use of pass1 stream)
      v8f hr = {}, hz = {}, hn = {};
      gemm3(WhT1, AH, cb, ln, hiw, hr, hz, hn);

      const v2f er = epi1g[0 * 128 + cb + ln];
      const v2f ez = epi1g[1 * 128 + cb + ln];
      const v2f en = epi1g[2 * 128 + cb + ln];
      #pragma unroll
      for (int g2 = 0; g2 < 8; g2 += 2) {
        const v2f vr = (pair(ir, g2) + pair(hr, g2)) + splat2(er.x + er.y);
        const v2f vz = (pair(iz, g2) + pair(hz, g2)) + splat2(ez.x + ez.y);
        const v2f rr = sig2(vr);
        const v2f zz = sig2(vz);
        const v2f vn = pair(inn, g2) + splat2(en.x) + rr * (pair(hn, g2) + splat2(en.y));
        const v2f nv = tanh2(vn);
        const v2f hc = pair(h1a[nt], g2);
        const v2f hn2 = nv + zz * (hc - nv);
        h1a[nt][g2] = hn2.x; h1a[nt][g2 + 1] = hn2.y;
      }
    }
  }   // time

  // ---- write final hidden states: out[2][B][H][N], 4 consecutive rows per b128 ----
  #pragma unroll
  for (int nt = 0; nt < 8; ++nt)
    #pragma unroll
    for (int g4 = 0; g4 < 8; g4 += 4) {
      const int M = M0 + g4 + hiw * 8;       // 4 consecutive rows => contiguous sn
      const int bb = M >> 9, sn = M & (NSENS - 1);
      const int hidx = nt * 16 + ln;
      v4f o0, o1;
      #pragma unroll
      for (int j = 0; j < 4; ++j) { o0[j] = h0a[nt][g4 + j]; o1[j] = h1a[nt][g4 + j]; }
      *(v4f*)(out + ((size_t)bb * HID + hidx) * NSENS + sn) = o0;
      *(v4f*)(out + (((size_t)256 + bb) * HID + hidx) * NSENS + sn) = o1;
    }
}

extern "C" void kernel_launch(void* const* d_in, const int* in_sizes, int n_in,
                              void* d_out, int out_size, void* d_ws, size_t ws_size,
                              hipStream_t stream) {
  const float* x   = (const float*)d_in[0];
  const float* Wx0 = (const float*)d_in[1];
  const float* Wh0 = (const float*)d_in[2];
  const float* bx0 = (const float*)d_in[3];
  const float* bh0 = (const float*)d_in[4];
  const float* Wx1 = (const float*)d_in[5];
  const float* Wh1 = (const float*)d_in[6];
  const float* bx1 = (const float*)d_in[7];
  const float* bh1 = (const float*)d_in[8];
  float* out = (float*)d_out;

  char* ws = (char*)d_ws;
  __bf16* wT   = (__bf16*)(ws + WS_WT);
  v4f*    epi0 = (v4f*)(ws + WS_EPI0);
  v2f*    epi1 = (v2f*)(ws + WS_EPI1);

  (void)hipFuncSetAttribute((const void*)gru_fused_kernel,
                            hipFuncAttributeMaxDynamicSharedMemorySize, SMEM_BYTES);

  // 1) prep weights/tables into workspace (runs every call; deterministic)
  gru_prep_kernel<<<(G3 * HID + 255) / 256, 256, 0, stream>>>(
      Wx0, Wh0, bx0, bh0, Wx1, Wh1, bx1, bh1, wT, epi0, epi1);

  // 2) fused persistent GRU
  const int rows = 256 * NSENS;          // B * N = 131072 independent chains
  const int blocks = rows / 128;         // 1024 workgroups, 8 waves x 16 rows
  gru_fused_kernel<<<blocks, BDIM, SMEM_BYTES, stream>>>(x, wT, epi0, epi1, out);
}
